// RGN_87883620810796
// MI455X (gfx1250) — compile-verified, matrix operand order checked by reading, not run on hardware
//
#include <hip/hip_runtime.h>
#include <hip/hip_bf16.h>

// ---------------- problem constants ----------------
#define LSEQ   576
#define BATCH  16
#define HID    800
#define G4     3200      // 4*HID
#define NT_G   200       // G4/16 n-tiles
#define KT_H   25        // HID/32 k-tiles
#define KT_X   2         // 41 -> 64 padded, /32
#define KT_1   50        // 1600/32
#define ALPHA  60
#define NFRAG  24
#define FRAGL  72        // 3*LSEQ/NFRAG

typedef __attribute__((ext_vector_type(16))) __bf16 v16bf;
typedef __attribute__((ext_vector_type(8)))  float  v8f;

__device__ inline v8f wmma_bf16(v16bf a, v16bf b, v8f c) {
    return __builtin_amdgcn_wmma_f32_16x16x32_bf16(
        false, a, false, b, (short)0, c, false, false);
}

// 16-bit A/B fragment K map (ISA 7.12.2, 16-bit A-Matrix 16x32):
// lane<16 : e<8 -> K=e        ; e>=8 -> K=e+8  (16..23)
// lane>=16: e<8 -> K=e+8      ; e>=8 -> K=e+16 (24..31)
__device__ inline int kmap_e(int e, int lanehi) {
    return e + (e >= 8 ? 8 : 0) + (lanehi ? 8 : 0);
}
__device__ inline void kinv(int k, int& lanehi, int& e) {
    if (k < 8)       { lanehi = 0; e = k;      }
    else if (k < 16) { lanehi = 1; e = k - 8;  }
    else if (k < 24) { lanehi = 0; e = k - 8;  }
    else             { lanehi = 1; e = k - 16; }
}

// ---------------- weight pack: row-major (Nout,Kin) f32 -> B fragments bf16 ----
// frag f = nt*KT + kt, element idx = lane*16+e ; B[k][n] = W[n][k]
__global__ void pack_b_kernel(const float* __restrict__ W, __bf16* __restrict__ dst,
                              int Nreal, int Kreal, int KT, int NT) {
    long total = (long)NT * KT * 512;
    for (long idx = (long)blockIdx.x * blockDim.x + threadIdx.x; idx < total;
         idx += (long)gridDim.x * blockDim.x) {
        int e    = (int)(idx & 15);
        int lane = (int)((idx >> 4) & 31);
        long f   = idx >> 9;
        int kt = (int)(f % KT);
        int nt = (int)(f / KT);
        int k  = kmap_e(e, lane >> 4);
        int n  = nt * 16 + (lane & 15);
        int gk = kt * 32 + k;
        float v = 0.f;
        if (n < Nreal && gk < Kreal) v = W[(long)n * Kreal + gk];
        dst[idx] = (__bf16)v;
    }
}

__global__ void bias_combine(const float* a, const float* b, float* out, int n) {
    int i = blockIdx.x * blockDim.x + threadIdx.x;
    if (i < n) out[i] = a[i] + b[i];
}

__global__ void init_ws(__bf16* hzero, unsigned* sync) {
    int i = blockIdx.x * blockDim.x + threadIdx.x;
    if (i < KT_H * 512) hzero[i] = (__bf16)0.f;
    if (i < 64) sync[i] = 0u;
}

// ---------------- x features -> A fragments (K padded 41->64) ----------------
__global__ void build_x(const int* __restrict__ seq, const float* __restrict__ pssm,
                        __bf16* __restrict__ xf) {
    int idx = blockIdx.x * blockDim.x + threadIdx.x;
    if (idx >= LSEQ * KT_X * 512) return;
    int e = idx & 15, lane = (idx >> 4) & 31;
    int f = idx >> 9;
    int kt = f & 1, t = f >> 1;
    int b = lane & 15;
    int gk = kt * 32 + kmap_e(e, lane >> 4);
    float v = 0.f;
    if (gk < 20)       v = (seq[t * BATCH + b] == gk) ? 1.f : 0.f;
    else if (gk < 41)  v = pssm[(t * BATCH + b) * 21 + (gk - 20)];
    xf[idx] = (__bf16)v;
}

// ---------------- persistent bidirectional LSTM layer -------------------------
// grid = 50 WGs: WG = dir*25 + hblk (32 hidden units per WG). 8 waves:
// wave w -> gate g=w/2, half=w%2 -> n-tile nt = g*50 + hblk*2 + half.
// h is carried between steps in WMMA A-fragment layout in global memory.
__global__ __launch_bounds__(256) void lstm_recurrent(
    const __bf16* __restrict__ whh_pack,   // [dir][200*25 frags]
    const __bf16* __restrict__ win_pack,   // layer0: [dir][200*2 frags]
    const __bf16* __restrict__ xfrag,      // layer0: [t][2 frags]
    const __bf16* __restrict__ precomp,    // layer1: [t][dir][nt][256] bf16
    const float*  __restrict__ bias,       // layer0: [dir][3200]
    __bf16* __restrict__ hout,             // [dir][t][25 frags]
    const __bf16* __restrict__ hzero,      // 25 zero frags
    unsigned* __restrict__ sync, int layer) {

    const int wg   = blockIdx.x;
    const int dir  = wg / 25;
    const int hblk = wg % 25;
    const int tid  = threadIdx.x;
    const int wave = tid >> 5, lane = tid & 31;
    const int g = wave >> 1, half = wave & 1;
    const int nt  = g * 50 + hblk * 2 + half;
    const int col = nt * 16 + (lane & 15);
    const int NWG = 50;

    __shared__ __align__(32) __bf16 sA[KT_H * 512];   // staged h fragments
    __shared__ float sG[4 * 32 * 16];                 // i,f,g,o for 32 hid x 16 batch

    float cst0 = 0.f, cst1 = 0.f;                     // cell state, 2 items/thread

    for (int s = 0; s < LSEQ; ++s) {
        int t = (dir == 0) ? s : (LSEQ - 1 - s);
        const __bf16* hprev = (s == 0)
            ? hzero
            : hout + ((size_t)(dir * LSEQ + (dir == 0 ? t - 1 : t + 1))) * (KT_H * 512);
        { // stage previous h fragments into LDS (6400 dwords)
            const uint32_t* src = (const uint32_t*)hprev;
            uint32_t* d = (uint32_t*)sA;
            for (int i = tid; i < KT_H * 512 / 2; i += 256) d[i] = src[i];
        }
        __syncthreads();

        v8f acc = {};
        if (layer == 0) {
            #pragma unroll
            for (int kt = 0; kt < KT_X; ++kt) {
                v16bf a = *(const v16bf*)(xfrag + ((size_t)t * KT_X + kt) * 512 + lane * 16);
                v16bf b = *(const v16bf*)(win_pack +
                          ((size_t)dir * (NT_G * KT_X) + (size_t)nt * KT_X + kt) * 512 + lane * 16);
                acc = wmma_bf16(a, b, acc);
            }
        }
        for (int kt = 0; kt < KT_H; ++kt) {
            v16bf a = *(const v16bf*)(&sA[kt * 512 + lane * 16]);
            v16bf b = *(const v16bf*)(whh_pack +
                      ((size_t)dir * (NT_G * KT_H) + (size_t)nt * KT_H + kt) * 512 + lane * 16);
            acc = wmma_bf16(a, b, acc);
        }

        const int nl = half * 16 + (lane & 15);       // local hidden index 0..31
        if (layer == 1) {
            const __bf16* pp = precomp +
                (((size_t)t * 2 + dir) * NT_G + nt) * 256 + lane * 8;
            #pragma unroll
            for (int r = 0; r < 8; ++r) {
                int m = r + 8 * (lane >> 4);
                sG[(g * 32 + nl) * 16 + m] = acc[r] + (float)pp[r];
            }
        } else {
            float bb = bias[dir * G4 + col];
            #pragma unroll
            for (int r = 0; r < 8; ++r) {
                int m = r + 8 * (lane >> 4);
                sG[(g * 32 + nl) * 16 + m] = acc[r] + bb;
            }
        }
        __syncthreads();

        // elementwise LSTM cell, 512 (hid,batch) pairs, 2 per thread
        float hv[2];
        #pragma unroll
        for (int rep = 0; rep < 2; ++rep) {
            int p = tid + 256 * rep;
            int hl = p >> 4, m = p & 15;
            float iv = sG[(0 * 32 + hl) * 16 + m];
            float fv = sG[(1 * 32 + hl) * 16 + m];
            float gv = sG[(2 * 32 + hl) * 16 + m];
            float ov = sG[(3 * 32 + hl) * 16 + m];
            float si = 1.f / (1.f + __expf(-iv));
            float sf = 1.f / (1.f + __expf(-fv));
            float so = 1.f / (1.f + __expf(-ov));
            float tg = tanhf(gv);
            float& cs = rep ? cst1 : cst0;
            cs = sf * cs + si * tg;
            hv[rep] = so * tanhf(cs);
        }
        // write h for this t in A-fragment layout (our k-tile == hblk)
        __bf16* hd = hout + ((size_t)(dir * LSEQ + t) * KT_H + hblk) * 512;
        #pragma unroll
        for (int rep = 0; rep < 2; ++rep) {
            int p = tid + 256 * rep;
            int k = p >> 4, m = p & 15;
            int lh, e; kinv(k, lh, e);
            hd[(m + 16 * lh) * 16 + e] = (__bf16)hv[rep];
        }
        __syncthreads();

        // grid-wide barrier (release h, acquire everyone else's h)
        __threadfence();
        __syncthreads();
        if (tid == 0) {
            atomicAdd(sync, 1u);
            unsigned target = (unsigned)(s + 1) * NWG;
            while (__hip_atomic_load(sync, __ATOMIC_ACQUIRE, __HIP_MEMORY_SCOPE_AGENT) < target)
                __builtin_amdgcn_s_sleep(1);
        }
        __syncthreads();
    }
}

// ---------------- layer-1 input GEMM: concat(h0f,h0b)(9216x1600) @ Wih1^T -----
// grid = LSEQ*2*25, each wave: one (t, 16-col) tile over K=1600 (50 k-iters)
__global__ __launch_bounds__(256) void precomp_l1(
    const __bf16* __restrict__ h0, const __bf16* __restrict__ wih1,
    const float* __restrict__ bias1, __bf16* __restrict__ out) {
    int wg  = blockIdx.x;
    int grp = wg % 25;
    int dir = (wg / 25) & 1;
    int t   = wg / 50;
    int tid = threadIdx.x, wave = tid >> 5, lane = tid & 31;
    int nt  = grp * 8 + wave;

    v8f acc = {};
    for (int kt = 0; kt < KT_1; ++kt) {
        const __bf16* asrc = (kt < KT_H)
            ? h0 + ((size_t)(0 * LSEQ + t) * KT_H + kt) * 512
            : h0 + ((size_t)(1 * LSEQ + t) * KT_H + (kt - KT_H)) * 512;
        v16bf a = *(const v16bf*)(asrc + lane * 16);
        v16bf b = *(const v16bf*)(wih1 +
                  ((size_t)dir * (NT_G * KT_1) + (size_t)nt * KT_1 + kt) * 512 + lane * 16);
        acc = wmma_bf16(a, b, acc);
    }
    float bb = bias1[dir * G4 + nt * 16 + (lane & 15)];
    __bf16* dst = out + (((size_t)t * 2 + dir) * NT_G + nt) * 256 + lane * 8;
    #pragma unroll
    for (int r = 0; r < 8; ++r) dst[r] = (__bf16)(acc[r] + bb);
}

// ---------------- FC + softmax + atan2 + torsion_to_srf ----------------------
__global__ __launch_bounds__(256) void head_kernel(
    const __bf16* __restrict__ h1, const __bf16* __restrict__ fcw,
    const float* __restrict__ fcb, const float* __restrict__ alphabet,
    const float* __restrict__ blen, const float* __restrict__ bang,
    float* __restrict__ srf) {
    int t = blockIdx.x;
    int tid = threadIdx.x, wave = tid >> 5, lane = tid & 31;
    __shared__ float logits[64 * 16];

    if (wave < 4) {   // N padded 60->64: 4 n-tiles
        int nt = wave;
        v8f acc = {};
        for (int kt = 0; kt < KT_1; ++kt) {
            const __bf16* asrc = (kt < KT_H)
                ? h1 + ((size_t)t * KT_H + kt) * 512
                : h1 + ((size_t)(LSEQ + t) * KT_H + (kt - KT_H)) * 512;
            v16bf a = *(const v16bf*)(asrc + lane * 16);
            v16bf b = *(const v16bf*)(fcw + ((size_t)nt * KT_1 + kt) * 512 + lane * 16);
            acc = wmma_bf16(a, b, acc);
        }
        int n = nt * 16 + (lane & 15);
        #pragma unroll
        for (int r = 0; r < 8; ++r) {
            int m = r + 8 * (lane >> 4);
            logits[n * 16 + m] = acc[r] + ((n < ALPHA) ? fcb[n] : 0.f);
        }
    }
    __syncthreads();

    if (tid < BATCH) {
        int b = tid;
        float mx = -1e30f;
        for (int a = 0; a < ALPHA; ++a) mx = fmaxf(mx, logits[a * 16 + b]);
        float pr[ALPHA]; float sum = 0.f;
        for (int a = 0; a < ALPHA; ++a) { pr[a] = __expf(logits[a * 16 + b] - mx); sum += pr[a]; }
        float inv = 1.f / sum;
        for (int j = 0; j < 3; ++j) {
            float ss = 0.f, cc = 0.f;
            for (int a = 0; a < ALPHA; ++a) {
                float p  = pr[a] * inv;
                float al = alphabet[a * 3 + j];
                ss += p * __sinf(al);
                cc += p * __cosf(al);
            }
            float phi = atan2f(ss, cc);
            float r = blen[j], th = bang[j];
            float* o = srf + ((size_t)(t * 3 + j) * BATCH + b) * 3;
            o[0] = r * __cosf(th);
            o[1] = r * __cosf(phi) * __sinf(th);
            o[2] = r * __sinf(phi) * __sinf(th);
        }
    }
}

// ---------------- pNeRF: 384-chain scan + 23 fragment merges -----------------
__device__ inline void frame3(float ax,float ay,float az, float bx,float by,float bz,
                              float cx,float cy,float cz,
                              float& bcx,float& bcy,float& bcz,
                              float& mx,float& my,float& mz,
                              float& nx,float& ny,float& nz) {
    bcx = cx - bx; bcy = cy - by; bcz = cz - bz;
    float il = rsqrtf(bcx*bcx + bcy*bcy + bcz*bcz);
    bcx *= il; bcy *= il; bcz *= il;
    float abx = bx - ax, aby = by - ay, abz = bz - az;
    nx = aby*bcz - abz*bcy; ny = abz*bcx - abx*bcz; nz = abx*bcy - aby*bcx;
    float inl = rsqrtf(nx*nx + ny*ny + nz*nz);
    nx *= inl; ny *= inl; nz *= inl;
    mx = ny*bcz - nz*bcy; my = nz*bcx - nx*bcz; mz = nx*bcy - ny*bcx;
}

__global__ __launch_bounds__(384) void pnerf_kernel(
    const float* __restrict__ srf, float* __restrict__ frag, float* __restrict__ out) {
    int tid = threadIdx.x;
    // phase A: scan 72 steps over 384 chains; chain = n*16 + b
    {
        int n = tid >> 4, b = tid & 15;
        float ax = -0.70710678f, ay = 1.22474487f, az = 0.f;
        float bx = -1.41421356f, by = 0.f, bz = 0.f;
        float cx = 0.f, cy = 0.f, cz = 0.f;
        for (int f = 0; f < FRAGL; ++f) {
            const float* ct = srf + ((size_t)(n * FRAGL + f) * BATCH + b) * 3;
            float bcx,bcy,bcz,mx,my,mz,nx,ny,nz;
            frame3(ax,ay,az, bx,by,bz, cx,cy,cz, bcx,bcy,bcz, mx,my,mz, nx,ny,nz);
            float dx = cx + bcx*ct[0] + mx*ct[1] + nx*ct[2];
            float dy = cy + bcy*ct[0] + my*ct[1] + ny*ct[2];
            float dz = cz + bcz*ct[0] + mz*ct[1] + nz*ct[2];
            float* fd = frag + ((size_t)f * 384 + tid) * 3;
            fd[0] = dx; fd[1] = dy; fd[2] = dz;
            ax = bx; ay = by; az = bz;
            bx = cx; by = cy; bz = cz;
            cx = dx; cy = dy; cz = dz;
        }
    }
    __threadfence(); __syncthreads();
    // coords[0:72] = frag[:, fragment 0]
    for (int i = tid; i < FRAGL * BATCH; i += 384) {
        int f = i >> 4, b = i & 15;
        const float* s = frag + ((size_t)f * 384 + b) * 3;
        float* o = out + ((size_t)f * BATCH + b) * 3;
        o[0] = s[0]; o[1] = s[1]; o[2] = s[2];
    }
    __threadfence(); __syncthreads();
    // merge fragments 1..23
    for (int k = 1; k < NFRAG; ++k) {
        int cur = FRAGL * k;
        for (int i = tid; i < FRAGL * BATCH; i += 384) {
            int f = i >> 4, b = i & 15;
            const float* pa = out + ((size_t)(cur - 3) * BATCH + b) * 3;
            const float* pb = out + ((size_t)(cur - 2) * BATCH + b) * 3;
            const float* pc = out + ((size_t)(cur - 1) * BATCH + b) * 3;
            float bcx,bcy,bcz,mx,my,mz,nx,ny,nz;
            frame3(pa[0],pa[1],pa[2], pb[0],pb[1],pb[2], pc[0],pc[1],pc[2],
                   bcx,bcy,bcz, mx,my,mz, nx,ny,nz);
            const float* fv = frag + ((size_t)f * 384 + k * 16 + b) * 3;
            float* o = out + ((size_t)(cur + f) * BATCH + b) * 3;
            o[0] = pc[0] + bcx*fv[0] + mx*fv[1] + nx*fv[2];
            o[1] = pc[1] + bcy*fv[0] + my*fv[1] + ny*fv[2];
            o[2] = pc[2] + bcz*fv[0] + mz*fv[1] + nz*fv[2];
        }
        __threadfence(); __syncthreads();
    }
}

// ---------------- host orchestration -----------------------------------------
extern "C" void kernel_launch(void* const* d_in, const int* in_sizes, int n_in,
                              void* d_out, int out_size, void* d_ws, size_t ws_size,
                              hipStream_t stream) {
    const int*   seq   = (const int*)  d_in[0];
    const float* pssm  = (const float*)d_in[1];
    const float* wih0f = (const float*)d_in[3],  *whh0f = (const float*)d_in[4];
    const float* bih0f = (const float*)d_in[5],  *bhh0f = (const float*)d_in[6];
    const float* wih0b = (const float*)d_in[7],  *whh0b = (const float*)d_in[8];
    const float* bih0b = (const float*)d_in[9],  *bhh0b = (const float*)d_in[10];
    const float* wih1f = (const float*)d_in[11], *whh1f = (const float*)d_in[12];
    const float* bih1f = (const float*)d_in[13], *bhh1f = (const float*)d_in[14];
    const float* wih1b = (const float*)d_in[15], *whh1b = (const float*)d_in[16];
    const float* bih1b = (const float*)d_in[17], *bhh1b = (const float*)d_in[18];
    const float* fcw   = (const float*)d_in[19], *fcb   = (const float*)d_in[20];
    const float* alpha = (const float*)d_in[21];
    const float* blen  = (const float*)d_in[22], *bang  = (const float*)d_in[23];
    (void)in_sizes; (void)n_in; (void)out_size; (void)ws_size;

    char* ws = (char*)d_ws; size_t off = 0;
    auto alloc = [&](size_t bytes) -> char* {
        char* p = ws + off; off = (off + bytes + 255) & ~(size_t)255; return p;
    };
    __bf16* pk_wih0 = (__bf16*)alloc((size_t)2 * NT_G * KT_X * 512 * 2);
    __bf16* pk_whh0 = (__bf16*)alloc((size_t)2 * NT_G * KT_H * 512 * 2);
    __bf16* pk_wih1 = (__bf16*)alloc((size_t)2 * NT_G * KT_1 * 512 * 2);
    __bf16* pk_whh1 = (__bf16*)alloc((size_t)2 * NT_G * KT_H * 512 * 2);
    __bf16* pk_fcw  = (__bf16*)alloc((size_t)4 * KT_1 * 512 * 2);
    float*  bias0   = (float*) alloc((size_t)2 * G4 * 4);
    float*  bias1   = (float*) alloc((size_t)2 * G4 * 4);
    __bf16* xfr     = (__bf16*)alloc((size_t)LSEQ * KT_X * 512 * 2);
    __bf16* h0      = (__bf16*)alloc((size_t)2 * LSEQ * KT_H * 512 * 2);
    __bf16* h1      = (__bf16*)alloc((size_t)2 * LSEQ * KT_H * 512 * 2);
    __bf16* hz      = (__bf16*)alloc((size_t)KT_H * 512 * 2);
    __bf16* pc1     = (__bf16*)alloc((size_t)2 * LSEQ * NT_G * 256 * 2);
    float*  srf     = (float*) alloc((size_t)3 * LSEQ * BATCH * 3 * 4);
    float*  fragb   = (float*) alloc((size_t)FRAGL * 384 * 3 * 4);
    unsigned* sync  = (unsigned*)alloc(256);

    init_ws<<<(KT_H * 512 + 255) / 256, 256, 0, stream>>>(hz, sync);

    auto packB = [&](const float* W, __bf16* dst, int Nreal, int Kreal, int KT, int NT) {
        long total = (long)NT * KT * 512;
        int grid = (int)((total + 255) / 256);
        pack_b_kernel<<<grid, 256, 0, stream>>>(W, dst, Nreal, Kreal, KT, NT);
    };
    packB(wih0f, pk_wih0,                        G4, 41,   KT_X, NT_G);
    packB(wih0b, pk_wih0 + (size_t)NT_G*KT_X*512,G4, 41,   KT_X, NT_G);
    packB(whh0f, pk_whh0,                        G4, HID,  KT_H, NT_G);
    packB(whh0b, pk_whh0 + (size_t)NT_G*KT_H*512,G4, HID,  KT_H, NT_G);
    packB(wih1f, pk_wih1,                        G4, 1600, KT_1, NT_G);
    packB(wih1b, pk_wih1 + (size_t)NT_G*KT_1*512,G4, 1600, KT_1, NT_G);
    packB(whh1f, pk_whh1,                        G4, HID,  KT_H, NT_G);
    packB(whh1b, pk_whh1 + (size_t)NT_G*KT_H*512,G4, HID,  KT_H, NT_G);
    packB(fcw,   pk_fcw,                         ALPHA, 1600, KT_1, 4);

    bias_combine<<<(G4 + 255) / 256, 256, 0, stream>>>(bih0f, bhh0f, bias0,      G4);
    bias_combine<<<(G4 + 255) / 256, 256, 0, stream>>>(bih0b, bhh0b, bias0 + G4, G4);
    bias_combine<<<(G4 + 255) / 256, 256, 0, stream>>>(bih1f, bhh1f, bias1,      G4);
    bias_combine<<<(G4 + 255) / 256, 256, 0, stream>>>(bih1b, bhh1b, bias1 + G4, G4);

    build_x<<<(LSEQ * KT_X * 512 + 255) / 256, 256, 0, stream>>>(seq, pssm, xfr);

    // layer 0: persistent bidirectional LSTM (50 WGs, grid barrier per step)
    lstm_recurrent<<<50, 256, 0, stream>>>(pk_whh0, pk_wih0, xfr, nullptr,
                                           bias0, h0, hz, sync, 0);
    // layer 1 input precompute (big parallel WMMA GEMM, biases folded in)
    precomp_l1<<<LSEQ * 2 * 25, 256, 0, stream>>>(h0, pk_wih1, bias1, pc1);
    // layer 1: persistent bidirectional LSTM
    lstm_recurrent<<<50, 256, 0, stream>>>(pk_whh1, nullptr, nullptr, pc1,
                                           nullptr, h1, hz, sync + 1, 1);
    // FC + softmax + angles + SRF
    head_kernel<<<LSEQ, 256, 0, stream>>>(h1, pk_fcw, fcb, alpha, blen, bang, srf);
    // pNeRF
    pnerf_kernel<<<1, 384, 0, stream>>>(srf, fragb, (float*)d_out);
}
